// GCN_13331578486813
// MI455X (gfx1250) — matrix operand order repaired and checked by reading, not compiled, not run on hardware
//
#include <hip/hip_runtime.h>

typedef __attribute__((ext_vector_type(16))) __bf16 v16bf;
typedef __attribute__((ext_vector_type(8)))  float  v8f;

// ---------------- degree / normalization ----------------

__global__ __launch_bounds__(256) void deg_init_k(float* __restrict__ deg, int n) {
  int i = blockIdx.x * 256 + threadIdx.x;
  if (i < n) deg[i] = 1.0f;  // self-loop contributes 1 to every node
}

__global__ __launch_bounds__(256) void deg_edges_k(const int* __restrict__ ei, int E,
                                                   float* __restrict__ deg) {
  int e = blockIdx.x * 256 + threadIdx.x;
  if (e < E) atomicAdd(&deg[ei[(long long)E + e]], 1.0f);
}

__global__ __launch_bounds__(256) void deg_rsqrt_k(float* __restrict__ deg, int n) {
  int i = blockIdx.x * 256 + threadIdx.x;
  if (i < n) deg[i] = rsqrtf(deg[i]);  // deg >= 1 always
}

// ---------------- weight pre-conversion (once, tiny) ----------------

// W1T bf16 [128 cols][512 k] (transposed W1): one contiguous B fragment per lane
__global__ __launch_bounds__(256) void cvtW1_k(const float* __restrict__ W1,
                                               __bf16* __restrict__ W1T) {
  int idx = blockIdx.x * 256 + threadIdx.x;  // 128*512 = 65536
  int c = idx >> 9, k = idx & 511;
  W1T[idx] = (__bf16)W1[(size_t)k * 128 + c];
}

// W2T bf16 [16 cols][128 k], columns 7..15 zero-padded
__global__ __launch_bounds__(256) void cvtW2_k(const float* __restrict__ W2,
                                               __bf16* __restrict__ W2T) {
  int idx = blockIdx.x * 256 + threadIdx.x;  // 16*128 = 2048
  int c = idx >> 7, k = idx & 127;
  W2T[idx] = (c < 7) ? (__bf16)W2[(size_t)k * 7 + c] : (__bf16)0.0f;
}

// ---------------- GEMM1: H[N,128] = bf16(X[N,512]) x W1T ----------------
// Each wave owns one 16-row M tile and computes ALL 8 N tiles, reusing the
// A fragment in registers across 8 WMMAs per k-step -> X is read exactly once.

__global__ __launch_bounds__(256) void gemm1_k(const float* __restrict__ X,
                                               const __bf16* __restrict__ W1T,
                                               float* __restrict__ H, int mtiles) {
  const int lane  = threadIdx.x & 31;
  const int wave  = threadIdx.x >> 5;
  const int mtile = blockIdx.x * 8 + wave;
  if (mtile >= mtiles) return;        // wave-uniform: EXEC all-ones for WMMA
  const int r16   = lane & 15;
  const int khalf = lane >> 4;        // which K half this lane holds
  const int row   = (mtile << 4) + r16;

  v8f acc[8] = {};
  const float* xrow = X + (size_t)row * 512;
  for (int k0 = 0; k0 < 512; k0 += 32) {
    // A 16x32 bf16: elems 0..15 -> K = k0+khalf*16+j
    v16bf a;
    const float* ap = xrow + k0 + khalf * 16;
#pragma unroll
    for (int j = 0; j < 8; ++j) {
      a[j]     = (__bf16)ap[j];
      a[j + 8] = (__bf16)ap[8 + j];
    }
#pragma unroll
    for (int n = 0; n < 8; ++n) {
      // B 32x16: lane holds column n*16+r16, K = k0+khalf*16+j  (contiguous 32B)
      v16bf b = *(const v16bf*)(W1T + (size_t)(n * 16 + r16) * 512 + k0 + khalf * 16);
      acc[n] = __builtin_amdgcn_wmma_f32_16x16x32_bf16(false, a, false, b, (short)0,
                                                       acc[n], false, false);
    }
  }
  // C/D layout: VGPR r holds row khalf*8 + r
#pragma unroll
  for (int n = 0; n < 8; ++n) {
    float* out = H + (size_t)((mtile << 4) + khalf * 8) * 128 + n * 16 + r16;
#pragma unroll
    for (int r = 0; r < 8; ++r) out[(size_t)r * 128] = acc[n][r];
  }
}

// ---------------- GEMM2: H2[N,7] = bf16(H1[N,128]) x W2T (N padded to 16) ----------------

__global__ __launch_bounds__(256) void gemm2_k(const float* __restrict__ H1,
                                               const __bf16* __restrict__ W2T,
                                               float* __restrict__ H2, int mtiles) {
  const int lane  = threadIdx.x & 31;
  const int wave  = threadIdx.x >> 5;
  const int mtile = blockIdx.x * 8 + wave;
  if (mtile >= mtiles) return;        // wave-uniform
  const int r16   = lane & 15;
  const int khalf = lane >> 4;
  const int row   = (mtile << 4) + r16;

  v8f acc = {};
  const float* hrow = H1 + (size_t)row * 128;
  for (int k0 = 0; k0 < 128; k0 += 32) {
    v16bf a;
    const float* ap = hrow + k0 + khalf * 16;
#pragma unroll
    for (int j = 0; j < 8; ++j) {
      a[j]     = (__bf16)ap[j];
      a[j + 8] = (__bf16)ap[8 + j];
    }
    v16bf b = *(const v16bf*)(W2T + (size_t)r16 * 128 + k0 + khalf * 16);
    acc = __builtin_amdgcn_wmma_f32_16x16x32_bf16(false, a, false, b, (short)0, acc,
                                                  false, false);
  }
  if (r16 < 7) {
    float* out = H2 + (size_t)((mtile << 4) + khalf * 8) * 7 + r16;
#pragma unroll
    for (int r = 0; r < 8; ++r) out[(size_t)r * 7] = acc[r];
  }
}

// ---------------- aggregation ----------------

// OUT[i,f] = H[i,f]*dinv[i]^2 + bias[f]   (self-loop message + bias folded in)
__global__ __launch_bounds__(256) void self_init_k(const float* __restrict__ H,
                                                   const float* __restrict__ dinv,
                                                   const float* __restrict__ bias,
                                                   float* __restrict__ OUT,
                                                   int n, int F) {
  long long idx = (long long)blockIdx.x * 256 + threadIdx.x;
  if (idx >= (long long)n * F) return;
  int i = (int)(idx / F), f = (int)(idx % F);
  float di = dinv[i];
  OUT[idx] = H[idx] * di * di + bias[f];
}

// 128-feature edge scatter: 32 lanes per edge, float4 each (L2-resident accumulator)
__global__ __launch_bounds__(256) void scatter128_k(const float* __restrict__ H,
                                                    const int* __restrict__ ei,
                                                    const float* __restrict__ dinv,
                                                    float* __restrict__ OUT, int E) {
  long long gid = (long long)blockIdx.x * 256 + threadIdx.x;
  int t = (int)(gid & 31);
  long long e = gid >> 5;
  if (e >= E) return;
  int src = ei[e];
  int dst = ei[(long long)E + e];
  float norm = dinv[src] * dinv[dst];
  float4 v = ((const float4*)(H + (size_t)src * 128))[t];
  float* o = OUT + (size_t)dst * 128 + t * 4;
  atomicAdd(o + 0, v.x * norm);
  atomicAdd(o + 1, v.y * norm);
  atomicAdd(o + 2, v.z * norm);
  atomicAdd(o + 3, v.w * norm);
}

// 7-feature edge scatter: one thread per edge
__global__ __launch_bounds__(256) void scatter7_k(const float* __restrict__ H,
                                                  const int* __restrict__ ei,
                                                  const float* __restrict__ dinv,
                                                  float* __restrict__ OUT, int E) {
  int e = blockIdx.x * 256 + threadIdx.x;
  if (e >= E) return;
  int src = ei[e];
  int dst = ei[(long long)E + e];
  float norm = dinv[src] * dinv[dst];
#pragma unroll
  for (int f = 0; f < 7; ++f)
    atomicAdd(&OUT[(size_t)dst * 7 + f], H[(size_t)src * 7 + f] * norm);
}

__global__ __launch_bounds__(256) void relu_k(const float* __restrict__ IN,
                                              float* __restrict__ OUT, long long total) {
  long long i = (long long)blockIdx.x * 256 + threadIdx.x;
  if (i < total) {
    float v = IN[i];
    OUT[i] = v > 0.0f ? v : 0.0f;
  }
}

// ---------------- driver ----------------

extern "C" void kernel_launch(void* const* d_in, const int* in_sizes, int n_in,
                              void* d_out, int out_size, void* d_ws, size_t ws_size,
                              hipStream_t stream) {
  const float* X  = (const float*)d_in[0];
  const int*   EI = (const int*)d_in[1];    // [2, E] flat: row 0 = src, row 1 = dst
  const float* W1 = (const float*)d_in[2];
  const float* B1 = (const float*)d_in[3];
  const float* W2 = (const float*)d_in[4];
  const float* B2 = (const float*)d_in[5];
  float* OUT = (float*)d_out;

  const int N = in_sizes[0] / 512;
  const int E = in_sizes[1] / 2;
  const int mtiles = N / 16;

  // workspace partition (~106 MB); all offsets stay 32B-aligned
  char* w = (char*)d_ws;
  float*  bufH   = (float*)w;   w += (size_t)N * 128 * sizeof(float);  // pre-agg h, then h1
  float*  bufAgg = (float*)w;   w += (size_t)N * 128 * sizeof(float);  // layer-1 accumulator
  float*  bufH2  = (float*)w;   w += (size_t)N * 7   * sizeof(float);  // pre-agg layer 2
  float*  dinv   = (float*)w;   w += (size_t)N       * sizeof(float);  // deg -> 1/sqrt(deg)
  __bf16* W1T    = (__bf16*)w;  w += (size_t)128 * 512 * sizeof(__bf16);
  __bf16* W2T    = (__bf16*)w;

  const int gN = (N + 255) / 256;
  const int gE = (E + 255) / 256;

  // normalization + weight conversion
  deg_init_k <<<gN, 256, 0, stream>>>(dinv, N);
  deg_edges_k<<<gE, 256, 0, stream>>>(EI, E, dinv);
  deg_rsqrt_k<<<gN, 256, 0, stream>>>(dinv, N);
  cvtW1_k<<<(128 * 512) / 256, 256, 0, stream>>>(W1, W1T);
  cvtW2_k<<<(16 * 128) / 256, 256, 0, stream>>>(W2, W2T);

  // layer 1
  gemm1_k<<<(mtiles + 7) / 8, 256, 0, stream>>>(X, W1T, bufH, mtiles);
  long long t1 = (long long)N * 128;
  self_init_k<<<(unsigned)((t1 + 255) / 256), 256, 0, stream>>>(bufH, dinv, B1, bufAgg, N, 128);
  scatter128_k<<<(unsigned)(((long long)E * 32 + 255) / 256), 256, 0, stream>>>(bufH, EI, dinv, bufAgg, E);
  relu_k<<<(unsigned)((t1 + 255) / 256), 256, 0, stream>>>(bufAgg, bufH, t1);

  // layer 2
  gemm2_k<<<(mtiles + 7) / 8, 256, 0, stream>>>(bufH, W2T, bufH2, mtiles);
  long long t2 = (long long)N * 7;
  self_init_k<<<(unsigned)((t2 + 255) / 256), 256, 0, stream>>>(bufH2, dinv, B2, OUT, N, 7);
  scatter7_k<<<gE, 256, 0, stream>>>(bufH2, EI, dinv, OUT, E);
}
